// butterflyAutoEncodeFre2_62981400428632
// MI455X (gfx1250) — matrix operand order repaired and checked by reading, not compiled, not run on hardware
//
#include <hip/hip_runtime.h>

typedef __attribute__((ext_vector_type(16))) _Float16 v16h;
typedef __attribute__((ext_vector_type(8)))  _Float16 v8h;
typedef __attribute__((ext_vector_type(4)))  _Float16 v4h;
typedef __attribute__((ext_vector_type(8)))  float    v8f;

#define BATCH    32768
#define DMODEL   1024
#define KBIG     1792
#define KSMALL   256
#define TM       16          // batch rows per block == WMMA M
#define NTHREADS 256         // 8 wave32

// ---- LDS layout (dynamic) ----
#define X_STRIDE  1032       // halfs (1024 + 8 pad, mult of 8 -> b128 aligned rows)
#define HA_STRIDE 1800       // halfs (1792 + 8 pad)
#define EA_STRIDE 264        // halfs (256 + 8 pad)
#define SM_X   0                                   // X16 [TM][X_STRIDE] f16 (33024 B)
#define SM_Y   (TM * X_STRIDE * 2)                 // Y1s [TM][X_STRIDE] f16 (33024 B)
#define SM_HA  (SM_Y + TM * X_STRIDE * 2)          // hA  [TM][HA_STRIDE] f16 (57600 B)
#define SM_EA  0                                   // eA aliases dead X16 region (8448 B)
#define SM_TOT (SM_HA + TM * HA_STRIDE * 2)        // 123648 B

// ---- workspace layout (halfs) ----
#define WS_ENC 0                                   // encW16: 256 x 1792
#define WS_DEC (KBIG * KSMALL)                     // decW16: 1024 x 256
#define WS_BM  (WS_DEC + DMODEL * KSMALL)          // bmat: 128 matrices of 32x32

__device__ __forceinline__ v16h combine16(v8h lo, v8h hi) {
    return __builtin_shufflevector(lo, hi, 0,1,2,3,4,5,6,7,8,9,10,11,12,13,14,15);
}

__device__ __forceinline__ v8f wmma_f16(v16h a, v16h b, v8f c) {
    return __builtin_amdgcn_wmma_f32_16x16x32_f16(
        false, a, false, b, (short)0, c, false, false);
}

// ======== kernel 0: fp32 -> fp16 weight conversion ========
__global__ void cvt_weights_kernel(const float* __restrict__ encw,
                                   const float* __restrict__ decw,
                                   _Float16* __restrict__ enc16,
                                   _Float16* __restrict__ dec16) {
    int i = blockIdx.x * blockDim.x + threadIdx.x;
    if (i < KBIG * KSMALL)   enc16[i] = (_Float16)encw[i];
    if (i < DMODEL * KSMALL) dec16[i] = (_Float16)decw[i];
}

// ======== kernel 1: collapse 5 butterfly stages into dense 32x32 f16 matrices ====
// mat = (st*2 + pass)*32 + c ; output Mws[mat][p][q] = combined M, row p, col q.
// pass 0: global stages 0..4 on aligned chunk c   (twiddle j = 16*c + jl)
// pass 1: global stages 5..9 on stride-32 chunk c (twiddle j = c + 32*jl)
__global__ void build_bfly_mats_kernel(const float* __restrict__ twiddle,
                                       _Float16* __restrict__ Mws) {
    const int mat  = blockIdx.x;          // 0..127
    const int c    = mat & 31;
    const int pass = (mat >> 5) & 1;
    const int st   = mat >> 6;
    const int q    = threadIdx.x;         // column, 0..31 (one wave per block)

    __shared__ float col[32][33];
    #pragma unroll
    for (int p = 0; p < 32; ++p) col[q][p] = (p == q) ? 1.0f : 0.0f;

    for (int sl = 0; sl < 5; ++sl) {
        const int s = 1 << sl;
        const int g = pass ? (5 + sl) : sl;          // global stage
        for (int jl = 0; jl < 16; ++jl) {
            const int p0  = ((jl >> sl) << (sl + 1)) | (jl & (s - 1));
            const int p1  = p0 + s;
            const int twj = pass ? (c + 32 * jl) : (16 * c + jl);
            const float4 t =
                ((const float4*)twiddle)[((size_t)st * 10 + g) * 512 + twj];
            float a = col[q][p0], b = col[q][p1];
            col[q][p0] = fmaf(t.x, a, t.y * b);
            col[q][p1] = fmaf(t.z, a, t.w * b);
        }
    }
    _Float16* outm = Mws + (size_t)mat * 1024;
    #pragma unroll
    for (int p = 0; p < 32; ++p) outm[p * 32 + q] = (_Float16)col[q][p];
}

// ======== kernel 2: fused butterfly(WMMA) + encode GEMM + decode GEMM ========
__global__ __launch_bounds__(NTHREADS)
void fused_bae_kernel(const float* __restrict__ x,
                      const float* __restrict__ enc_b,
                      const float* __restrict__ dec_b,
                      const _Float16* __restrict__ encW,   // (256,1792) f16
                      const _Float16* __restrict__ decW,   // (1024,256) f16
                      const _Float16* __restrict__ bmat,   // 128 x 32x32 f16
                      float* __restrict__ out)             // (B,1024) f32
{
    extern __shared__ char smem[];
    _Float16* X16 = (_Float16*)(smem + SM_X);
    _Float16* Y1s = (_Float16*)(smem + SM_Y);
    _Float16* hA  = (_Float16*)(smem + SM_HA);
    _Float16* eA  = (_Float16*)(smem + SM_EA);   // aliases X16 (dead by then)

    const int tid  = threadIdx.x;
    const int row0 = blockIdx.x * TM;
    const int wave  = tid >> 5;
    const int lane  = tid & 31;
    const int laneM = lane & 15;
    const int laneH = lane >> 4;

    // ---- load x tile (16x1024), convert to f16 once (shared by both stacks) ----
    for (int i = tid; i < TM * (DMODEL / 4); i += NTHREADS) {
        int r = i / (DMODEL / 4), c4 = i % (DMODEL / 4);
        float4 v = ((const float4*)x)[(size_t)(row0 + r) * (DMODEL / 4) + c4];
        v4h hv = { (_Float16)v.x, (_Float16)v.y, (_Float16)v.z, (_Float16)v.w };
        *(v4h*)&X16[r * X_STRIDE + c4 * 4] = hv;
    }
    __syncthreads();

    // ================= Phase A: butterfly as 2 block-diagonal WMMA passes =======
    for (int st = 0; st < 2; ++st) {
        // pass 1: Y1[.,32c+n] = sum_q X[.,32c+q] * M1[c][n][q], stored bit-swapped:
        //         Y1s[m][n*32 + c]
        for (int c = wave; c < 32; c += 8) {
            const _Float16* Arow = X16 + laneM * X_STRIDE + 32 * c + 16 * laneH;
            v16h a = combine16(*(const v8h*)Arow, *(const v8h*)(Arow + 8));
            const _Float16* Mb = bmat + ((size_t)((st * 2 + 0) * 32 + c)) * 1024;
            const _Float16* B0 = Mb + (size_t)laneM * 32 + 16 * laneH;
            const _Float16* B1 = Mb + (size_t)(laneM + 16) * 32 + 16 * laneH;
            v16h b0 = combine16(*(const v8h*)B0, *(const v8h*)(B0 + 8));
            v16h b1 = combine16(*(const v8h*)B1, *(const v8h*)(B1 + 8));
            v8f z = {};
            v8f d0 = wmma_f16(a, b0, z);
            v8f d1 = wmma_f16(a, b1, z);
            #pragma unroll
            for (int v = 0; v < 8; ++v) {
                const int m = v + 8 * laneH;
                Y1s[m * X_STRIDE + laneM * 32 + c]        = (_Float16)d0[v];
                Y1s[m * X_STRIDE + (laneM + 16) * 32 + c] = (_Float16)d1[v];
            }
        }
        __syncthreads();

        // pass 2: out col pos = c1 + 32*i ; A rows are contiguous (bit-swapped) runs
        //         hA[m][st*1024 + c1 + 32*i] = sum_j Y1s[m][c1*32+j] * M2[c1][i][j]
        for (int c1 = wave; c1 < 32; c1 += 8) {
            const _Float16* Arow = Y1s + laneM * X_STRIDE + 32 * c1 + 16 * laneH;
            v16h a = combine16(*(const v8h*)Arow, *(const v8h*)(Arow + 8));
            const _Float16* Mb = bmat + ((size_t)((st * 2 + 1) * 32 + c1)) * 1024;
            const _Float16* B0 = Mb + (size_t)laneM * 32 + 16 * laneH;
            const _Float16* B1 = Mb + (size_t)(laneM + 16) * 32 + 16 * laneH;
            v16h b0 = combine16(*(const v8h*)B0, *(const v8h*)(B0 + 8));
            v16h b1 = combine16(*(const v8h*)B1, *(const v8h*)(B1 + 8));
            v8f z = {};
            v8f d0 = wmma_f16(a, b0, z);
            v8f d1 = wmma_f16(a, b1, z);
            const int cb = st * DMODEL;
            #pragma unroll
            for (int v = 0; v < 8; ++v) {
                const int m = v + 8 * laneH;
                // i = laneM: col <= 31+32*15 = 511 (+cb) -> always kept
                hA[m * HA_STRIDE + cb + c1 + 32 * laneM] = (_Float16)d0[v];
                // i = laneM+16: stack1 keeps only cols < 768 (i <= 23)
                if (st == 0 || laneM < 8)
                    hA[m * HA_STRIDE + cb + c1 + 32 * (laneM + 16)] = (_Float16)d1[v];
            }
        }
        __syncthreads();
    }

    // ================= Phase B: encode  e(16x256) = hA(16x1792) @ encW^T =========
    {
        const int n0 = wave * 32;
        v8f c0 = {}, c1 = {};
        const _Float16* W0   = encW + (size_t)(n0 + laneM) * KBIG;
        const _Float16* W1   = encW + (size_t)(n0 + 16 + laneM) * KBIG;
        const _Float16* Arow = hA + laneM * HA_STRIDE + 16 * laneH;
        for (int k0 = 0; k0 < KBIG; k0 += 32) {
            v16h a = combine16(*(const v8h*)(Arow + k0), *(const v8h*)(Arow + k0 + 8));
            const int kb = k0 + 16 * laneH;
            v16h b0 = combine16(*(const v8h*)(W0 + kb), *(const v8h*)(W0 + kb + 8));
            v16h b1 = combine16(*(const v8h*)(W1 + kb), *(const v8h*)(W1 + kb + 8));
            c0 = wmma_f16(a, b0, c0);
            c1 = wmma_f16(a, b1, c1);
        }
        const float bias0 = enc_b[n0 + laneM];
        const float bias1 = enc_b[n0 + 16 + laneM];
        #pragma unroll
        for (int v = 0; v < 8; ++v) {
            const int m = v + 8 * laneH;
            eA[m * EA_STRIDE + n0 + laneM]      = (_Float16)(c0[v] + bias0);
            eA[m * EA_STRIDE + n0 + 16 + laneM] = (_Float16)(c1[v] + bias1);
        }
    }
    __syncthreads();

    // ================= Phase C: decode  out(16x1024) = eA(16x256) @ decW^T =======
    {
        const int n0 = wave * 128;
        v8f zero = {};
        v8f acc[8];
        #pragma unroll
        for (int f = 0; f < 8; ++f) acc[f] = zero;

        const _Float16* Erow = eA + laneM * EA_STRIDE + 16 * laneH;
        for (int k0 = 0; k0 < KSMALL; k0 += 32) {
            v16h a = combine16(*(const v8h*)(Erow + k0), *(const v8h*)(Erow + k0 + 8));
            #pragma unroll
            for (int f = 0; f < 8; ++f) {
                const _Float16* Wp =
                    decW + (size_t)(n0 + f * 16 + laneM) * KSMALL + k0 + 16 * laneH;
                v16h b = combine16(*(const v8h*)Wp, *(const v8h*)(Wp + 8));
                acc[f] = wmma_f16(a, b, acc[f]);
            }
        }
        #pragma unroll
        for (int f = 0; f < 8; ++f) {
            const int n = n0 + f * 16 + laneM;
            const float bias = dec_b[n];
            #pragma unroll
            for (int v = 0; v < 8; ++v) {
                const int m = v + 8 * laneH;
                out[(size_t)(row0 + m) * DMODEL + n] = acc[f][v] + bias;
            }
        }
    }
}

extern "C" void kernel_launch(void* const* d_in, const int* in_sizes, int n_in,
                              void* d_out, int out_size, void* d_ws, size_t ws_size,
                              hipStream_t stream) {
    const float* x  = (const float*)d_in[0];
    const float* tw = (const float*)d_in[1];
    const float* ew = (const float*)d_in[2];
    const float* eb = (const float*)d_in[3];
    const float* dw = (const float*)d_in[4];
    const float* db = (const float*)d_in[5];
    float* out = (float*)d_out;

    _Float16* enc16 = (_Float16*)d_ws + WS_ENC;
    _Float16* dec16 = (_Float16*)d_ws + WS_DEC;
    _Float16* bm16  = (_Float16*)d_ws + WS_BM;

    hipLaunchKernelGGL(build_bfly_mats_kernel, dim3(128), dim3(32), 0, stream,
                       tw, bm16);
    hipLaunchKernelGGL(cvt_weights_kernel,
                       dim3((KBIG * KSMALL + 255) / 256), dim3(256), 0, stream,
                       ew, dw, enc16, dec16);
    hipLaunchKernelGGL(fused_bae_kernel,
                       dim3(BATCH / TM), dim3(NTHREADS), SM_TOT, stream,
                       x, eb, db, enc16, dec16, bm16, out);
}
